// GatingNetworkWithTopK_84765474554319
// MI455X (gfx1250) — matrix operand order repaired and checked by reading, not compiled, not run on hardware
//
#include <hip/hip_runtime.h>

#define N_TOK 32768
#define D_IN  1024
#define H_DIM 2048
#define N_EXP 16

// fused-kernel tiling
#define BM 128                 // tokens per block (8 M-tiles of 16)
#define BH 256                 // hidden columns per outer tile (16 N-tiles)
#define KT 32                  // K chunk per staged tile (one WMMA K-step)
#define NKK (D_IN / KT)        // 32 tiles per hidden sweep
#define XS_STRIDE 40           // bf16 elems per xs row (32 + 8 pad)
#define W1S_STRIDE 40
#define HB_STRIDE 136          // bf16 elems per h-staging row (128 + 8 pad)

#define XS_BYTES   (128 * XS_STRIDE * 2)    // 10240
#define W1S_BYTES  (256 * W1S_STRIDE * 2)   // 20480
#define BUF_BYTES  (XS_BYTES + W1S_BYTES)   // 30720 per double-buffer slot
#define HB_WBYTES  4352                     // 16*136*2 per-wave h staging
#define LRED_OFF   34816                    // after hbuf[8] (union region)
#define SMEM_BYTES (2 * BUF_BYTES)          // 61440

typedef unsigned int u32;
typedef __attribute__((ext_vector_type(2))) u32   u32x2;
typedef __attribute__((ext_vector_type(4))) u32   u32x4;
typedef __attribute__((ext_vector_type(4))) int   v4i;
typedef __attribute__((ext_vector_type(4))) float f32x4;
typedef __attribute__((ext_vector_type(16))) __bf16 v16bf;
typedef __attribute__((ext_vector_type(8)))  float  v8f;

union Frag32 { struct { u32x4 lo, hi; } q; v16bf v; };

__device__ __forceinline__ unsigned short f2bf(float f) {
  u32 u = __float_as_uint(f);
  u += 0x7FFFu + ((u >> 16) & 1u);
  return (unsigned short)(u >> 16);
}

#if __has_builtin(__builtin_amdgcn_cvt_pk_bf16_f32)
__device__ __forceinline__ u32 pack_bf16(float a, float b) {
  auto p = __builtin_amdgcn_cvt_pk_bf16_f32(a, b);   // v_cvt_pk_bf16_f32
  u32 r;
  __builtin_memcpy(&r, &p, 4);
  return r;
}
__device__ __forceinline__ unsigned short f2bf1(float f) {
  return (unsigned short)pack_bf16(f, f);
}
#else
__device__ __forceinline__ u32 pack_bf16(float a, float b) {
  return (u32)f2bf(a) | ((u32)f2bf(b) << 16);
}
__device__ __forceinline__ unsigned short f2bf1(float f) { return f2bf(f); }
#endif

// async global->LDS 16B copy (CDNA5 GLOBAL_LOAD_ASYNC_TO_LDS_B128, ASYNCcnt)
#if __has_builtin(__builtin_amdgcn_global_load_async_to_lds_b128)
__device__ __forceinline__ void async_cp16(const unsigned short* g, unsigned short* l) {
  __builtin_amdgcn_global_load_async_to_lds_b128(
      (__attribute__((address_space(1))) v4i*)(unsigned short*)g,
      (__attribute__((address_space(3))) v4i*)l, 0, 0);
}
#else
__device__ __forceinline__ void async_cp16(const unsigned short* g, unsigned short* l) {
  *(u32x4*)l = *(const u32x4*)g;   // sync fallback (ASYNCcnt stays 0; waits no-op)
}
#endif

template <int N>
__device__ __forceinline__ void wait_async_n() {
#if __has_builtin(__builtin_amdgcn_s_wait_asynccnt)
  __builtin_amdgcn_s_wait_asynccnt(N);
#else
  asm volatile("s_wait_asynccnt %0" :: "i"(N) : "memory");
#endif
}

__device__ __forceinline__ v16bf ld_frag(const unsigned short* p) {
  Frag32 f;
  f.q.lo = *(const u32x4*)(p);
  f.q.hi = *(const u32x4*)(p + 8);
  return f.v;
}

// ---------------------------------------------------------------- prep: W1T
// W1[K=1024][N=2048] fp32 -> W1T[N=2048][K=1024] bf16 (tiled 32x32 transpose)
__global__ __launch_bounds__(256) void prep_w1t(const float* __restrict__ W1,
                                                unsigned short* __restrict__ w1t) {
  __shared__ unsigned short tile[32][33];
  const int bk = blockIdx.x & 31;
  const int bn = blockIdx.x >> 5;
  const int tid = threadIdx.x;
#pragma unroll
  for (int j = 0; j < 4; ++j) {
    int idx = j * 256 + tid;
    int kr = idx >> 5, nc = idx & 31;
    tile[kr][nc] = f2bf1(W1[(size_t)(bk * 32 + kr) * H_DIM + bn * 32 + nc]);
  }
  __syncthreads();
#pragma unroll
  for (int j = 0; j < 4; ++j) {
    int idx = j * 256 + tid;
    int nr = idx >> 5, kc = idx & 31;
    w1t[(size_t)(bn * 32 + nr) * D_IN + bk * 32 + kc] = tile[kc][nr];
  }
}

// ---------------------------------------------------------------- prep: W2T
// W2[H=2048][E=16] fp32 -> W2T[E=16][H=2048] bf16
__global__ __launch_bounds__(256) void prep_w2t(const float* __restrict__ W2,
                                                unsigned short* __restrict__ w2t) {
  int i = blockIdx.x * 256 + threadIdx.x;
  int h = i >> 4, e = i & 15;
  w2t[(size_t)e * H_DIM + h] = f2bf1(W2[i]);
}

// ---------------------------------------------------------------- prep: xbf
// one-pass fp32 -> bf16 of x so hot-loop staging is a pure async copy
__global__ __launch_bounds__(256) void prep_xbf(const float* __restrict__ x,
                                                unsigned short* __restrict__ xbf) {
  size_t i = ((size_t)blockIdx.x * 256 + threadIdx.x) * 8;
  f32x4 a = *(const f32x4*)(x + i);
  f32x4 b = *(const f32x4*)(x + i + 4);
  u32x4 p;
  p.x = pack_bf16(a.x, a.y);
  p.y = pack_bf16(a.z, a.w);
  p.z = pack_bf16(b.x, b.y);
  p.w = pack_bf16(b.z, b.w);
  *(u32x4*)(xbf + i) = p;
}

// ----------------------------------------------------------- staging helper
template <bool PRECONV>
__device__ __forceinline__ void stage_tile(unsigned short* xsb, unsigned short* w1sb,
                                           const float* x, const unsigned short* xbf,
                                           const unsigned short* w1t,
                                           int row0, int hc, int kk, int tid) {
  if constexpr (PRECONV) {   // pure async bf16 copy of x tile [128][32]
#pragma unroll
    for (int it = 0; it < 2; ++it) {
      int linear = it * 256 + tid;
      int r = linear >> 2, c = linear & 3;
      async_cp16(xbf + (size_t)(row0 + r) * D_IN + kk + c * 8,
                 xsb + r * XS_STRIDE + c * 8);
    }
  } else {                   // inline fp32->bf16 conversion fallback
#pragma unroll
    for (int it = 0; it < 4; ++it) {
      int linear = it * 256 + tid;
      int r = linear >> 3, c4 = linear & 7;
      const float* gp = x + (size_t)(row0 + r) * D_IN + kk + c4 * 4;
      if (kk + KT < D_IN) __builtin_prefetch(gp + KT, 0, 1);
      f32x4 v = *(const f32x4*)gp;
      u32x2 pk;
      pk.x = pack_bf16(v.x, v.y);
      pk.y = pack_bf16(v.z, v.w);
      *(u32x2*)(xsb + r * XS_STRIDE + c4 * 4) = pk;
    }
  }
  // W1T tile [256][32] bf16, async copy
#pragma unroll
  for (int it = 0; it < 4; ++it) {
    int linear = it * 256 + tid;
    int nrow = linear >> 2, c = linear & 3;
    async_cp16(w1t + (size_t)(hc + nrow) * D_IN + kk + c * 8,
               w1sb + nrow * W1S_STRIDE + c * 8);
  }
}

// ------------------------------------------------------- fused gating kernel
// out[t][e] = softmax(relu(x@W1+b1)@W2+b2)[t][e] * onehot(argmax)
// 8 waves = 4 M-groups (2 M-tiles each) x 2 N-groups (8 N-tiles each)
template <bool PRECONV>
__global__ __launch_bounds__(256, 1) void moe_gate_fused(
    const float* __restrict__ x, const unsigned short* __restrict__ xbf,
    const unsigned short* __restrict__ w1t, const float* __restrict__ b1,
    const unsigned short* __restrict__ w2t, const float* __restrict__ b2,
    float* __restrict__ out) {
  __shared__ __align__(16) unsigned char smem[SMEM_BYTES];
  unsigned short* xs0  = (unsigned short*)(smem);
  unsigned short* w1s0 = (unsigned short*)(smem + XS_BYTES);
  unsigned short* xs1  = (unsigned short*)(smem + BUF_BYTES);
  unsigned short* w1s1 = (unsigned short*)(smem + BUF_BYTES + XS_BYTES);

  const int tid  = threadIdx.x;
  const int lane = tid & 31;
  const int wave = tid >> 5;
  const int mg   = wave & 3;
  const int ng   = wave >> 2;
  const int hl   = lane >> 4;
  const int l16  = lane & 15;
  const int row0 = blockIdx.x * BM;
  constexpr int OUTST = PRECONV ? 6 : 4;   // async ops in flight per tile

  unsigned short* hb = (unsigned short*)(smem + wave * HB_WBYTES);  // [16][136]

  v8f logits[2];
  logits[0] = (v8f){0.f,0.f,0.f,0.f,0.f,0.f,0.f,0.f};
  logits[1] = (v8f){0.f,0.f,0.f,0.f,0.f,0.f,0.f,0.f};

#pragma clang loop unroll(disable)
  for (int hc = 0; hc < H_DIM; hc += BH) {
    v8f acc[16];
#pragma unroll
    for (int t = 0; t < 16; ++t) acc[t] = (v8f){0.f,0.f,0.f,0.f,0.f,0.f,0.f,0.f};

    __syncthreads();  // separate prev GEMM2/epilogue LDS use from staging
    stage_tile<PRECONV>(xs0, w1s0, x, xbf, w1t, row0, hc, 0, tid);

#pragma clang loop unroll(disable)
    for (int kki = 0; kki < NKK; kki += 2) {
      // ---- phase A: prefetch tile kki+1 -> buf1, compute buf0 (tile kki)
      if (kki + 1 < NKK)
        stage_tile<PRECONV>(xs1, w1s1, x, xbf, w1t, row0, hc, (kki + 1) * KT, tid);
      wait_async_n<OUTST>();
      __syncthreads();
      {
        v16bf a0 = ld_frag(xs0 + (mg * 32 +      l16) * XS_STRIDE + hl * 16);
        v16bf a1 = ld_frag(xs0 + (mg * 32 + 16 + l16) * XS_STRIDE + hl * 16);
#pragma unroll
        for (int j = 0; j < 8; ++j) {
          v16bf b = ld_frag(w1s0 + (ng * 128 + j * 16 + l16) * W1S_STRIDE + hl * 16);
          acc[j]     = __builtin_amdgcn_wmma_f32_16x16x32_bf16(
              false, a0, false, b, (short)0, acc[j], false, false);
          acc[8 + j] = __builtin_amdgcn_wmma_f32_16x16x32_bf16(
              false, a1, false, b, (short)0, acc[8 + j], false, false);
        }
      }
      __syncthreads();

      // ---- phase B: prefetch tile kki+2 -> buf0, compute buf1 (tile kki+1)
      if (kki + 2 < NKK) {
        stage_tile<PRECONV>(xs0, w1s0, x, xbf, w1t, row0, hc, (kki + 2) * KT, tid);
        wait_async_n<OUTST>();
      } else {
        wait_async_n<0>();     // drain
      }
      __syncthreads();
      {
        v16bf a0 = ld_frag(xs1 + (mg * 32 +      l16) * XS_STRIDE + hl * 16);
        v16bf a1 = ld_frag(xs1 + (mg * 32 + 16 + l16) * XS_STRIDE + hl * 16);
#pragma unroll
        for (int j = 0; j < 8; ++j) {
          v16bf b = ld_frag(w1s1 + (ng * 128 + j * 16 + l16) * W1S_STRIDE + hl * 16);
          acc[j]     = __builtin_amdgcn_wmma_f32_16x16x32_bf16(
              false, a0, false, b, (short)0, acc[j], false, false);
          acc[8 + j] = __builtin_amdgcn_wmma_f32_16x16x32_bf16(
              false, a1, false, b, (short)0, acc[8 + j], false, false);
        }
      }
      __syncthreads();
    }

    // --- bias + relu -> bf16 h staging (wave-private rows), GEMM2 partials
#pragma unroll
    for (int mt = 0; mt < 2; ++mt) {
      asm volatile("" ::: "memory");
#pragma unroll
      for (int j = 0; j < 8; ++j) {
        float bias = b1[hc + ng * 128 + j * 16 + l16];
#pragma unroll
        for (int r = 0; r < 8; ++r) {
          float v = acc[mt * 8 + j][r] + bias;
          v = v > 0.f ? v : 0.f;
          hb[(r + 8 * hl) * HB_STRIDE + j * 16 + l16] = f2bf1(v);
        }
      }
      asm volatile("" ::: "memory");  // keep hb loads after hb stores
#pragma unroll
      for (int kb2 = 0; kb2 < 4; ++kb2) {
        v16bf af = ld_frag(hb + l16 * HB_STRIDE + kb2 * 32 + hl * 16);
        v16bf bf_ = ld_frag(w2t + (size_t)l16 * H_DIM + hc + ng * 128 +
                            kb2 * 32 + hl * 16);
        logits[mt] = __builtin_amdgcn_wmma_f32_16x16x32_bf16(
            false, af, false, bf_, (short)0, logits[mt], false, false);
      }
    }
  }

  // --- cross-N-group logits reduction (hidden dim split over ng)
  float* lred = (float*)(smem + LRED_OFF);
  __syncthreads();
  if (ng == 1) {
#pragma unroll
    for (int mt = 0; mt < 2; ++mt)
#pragma unroll
      for (int r = 0; r < 8; ++r)
        lred[(mg * 2 + mt) * 256 + r * 32 + lane] = logits[mt][r];
  }
  __syncthreads();

  // --- epilogue (ng==0): +b2, softmax over 16 experts, top-1 one-hot mask
  if (ng == 0) {
    float b2v = b2[l16];
#pragma unroll
    for (int mt = 0; mt < 2; ++mt) {
#pragma unroll
      for (int r = 0; r < 8; ++r) {
        float lv = logits[mt][r] + lred[(mg * 2 + mt) * 256 + r * 32 + lane] + b2v;
        float mx = lv;
#pragma unroll
        for (int m = 1; m <= 8; m <<= 1) mx = fmaxf(mx, __shfl_xor(mx, m, 32));
        float ex = __expf(lv - mx);
        float sm = ex;
#pragma unroll
        for (int m = 1; m <= 8; m <<= 1) sm += __shfl_xor(sm, m, 32);
        float gate = ex / sm;
        int cand = (lv == mx) ? l16 : N_EXP;   // tie -> lowest expert index
#pragma unroll
        for (int m = 1; m <= 8; m <<= 1) {
          int o = __shfl_xor(cand, m, 32);
          cand = o < cand ? o : cand;
        }
        float val = (l16 == cand) ? gate : 0.f;
        int token = row0 + mg * 32 + mt * 16 + r + 8 * hl;
        out[(size_t)token * N_EXP + l16] = val;
      }
    }
  }
}

// ----------------------------------------------- deterministic column reduce
__global__ __launch_bounds__(256) void col_reduce(const float* __restrict__ out,
                                                  float* __restrict__ sums) {
  __shared__ float red[256];
  const int e = blockIdx.x;
  float s = 0.f;
  for (int t = threadIdx.x; t < N_TOK; t += 256) s += out[(size_t)t * N_EXP + e];
  red[threadIdx.x] = s;
  __syncthreads();
  for (int off = 128; off > 0; off >>= 1) {
    if (threadIdx.x < off) red[threadIdx.x] += red[threadIdx.x + off];
    __syncthreads();
  }
  if (threadIdx.x == 0) sums[e] = red[0];
}

// ------------------------------------------------------------- final scaling
__global__ __launch_bounds__(256) void normalize_k(float* __restrict__ out,
                                                   const float* __restrict__ sums) {
  __shared__ float sc[16];
  if (threadIdx.x < 16) sc[threadIdx.x] = 32768.0f / (sums[threadIdx.x] + 1e-4f);
  __syncthreads();
  int i = blockIdx.x * 256 + threadIdx.x;
  out[i] = out[i] * sc[i & 15];
}

extern "C" void kernel_launch(void* const* d_in, const int* in_sizes, int n_in,
                              void* d_out, int out_size, void* d_ws, size_t ws_size,
                              hipStream_t stream) {
  (void)in_sizes; (void)n_in; (void)out_size;
  const float* x  = (const float*)d_in[0];
  const float* W1 = (const float*)d_in[1];
  const float* b1 = (const float*)d_in[2];
  const float* W2 = (const float*)d_in[3];
  const float* b2 = (const float*)d_in[4];
  float* out = (float*)d_out;

  // ws: W1T bf16 (4 MB) | W2T bf16 (64 KB) | sums | (optional) xbf bf16 (64 MB)
  size_t off_w2t  = (size_t)D_IN * H_DIM * 2;
  size_t off_sums = off_w2t + (size_t)H_DIM * N_EXP * 2;
  size_t off_xbf  = off_sums + 1024;
  size_t need     = off_xbf + (size_t)N_TOK * D_IN * 2;

  unsigned short* w1t = (unsigned short*)d_ws;
  unsigned short* w2t = (unsigned short*)((char*)d_ws + off_w2t);
  float* sums = (float*)((char*)d_ws + off_sums);
  unsigned short* xbf = (unsigned short*)((char*)d_ws + off_xbf);

  prep_w1t<<<(D_IN / 32) * (H_DIM / 32), 256, 0, stream>>>(W1, w1t);
  prep_w2t<<<(H_DIM * N_EXP) / 256, 256, 0, stream>>>(W2, w2t);
  if (ws_size >= need) {
    prep_xbf<<<(N_TOK * (D_IN / 8)) / 256, 256, 0, stream>>>(x, xbf);
    moe_gate_fused<true><<<N_TOK / BM, 256, 0, stream>>>(x, xbf, w1t, b1, w2t, b2, out);
  } else {
    moe_gate_fused<false><<<N_TOK / BM, 256, 0, stream>>>(x, xbf, w1t, b1, w2t, b2, out);
  }
  col_reduce<<<N_EXP, 256, 0, stream>>>(out, sums);
  normalize_k<<<(N_TOK * N_EXP) / 256, 256, 0, stream>>>(out, sums);
}
